// SpGraphAttentionLayer_71236327571610
// MI455X (gfx1250) — matrix operand order repaired and checked by reading, not compiled, not run on hardware
//
#include <hip/hip_runtime.h>
#include <hip/hip_bf16.h>

typedef __attribute__((ext_vector_type(8)))  float   v8f;
typedef __attribute__((ext_vector_type(8)))  __bf16  v8bf;
typedef __attribute__((ext_vector_type(16))) __bf16  v16bf;

#define N_NODES 50000
#define IN_F    128
#define OUT_F   128
#define NRELA   128
#define E1      400000
#define E2      100000
#define E_TOT   (E1 + E2)
#define K_TOT   (2 * IN_F + NRELA)      // 384
#define KCH     (K_TOT / 32)            // 12 K-chunks of 32
#define NCH     (OUT_F / 16)            // 8 N-chunks of 16
#define TILES   (E_TOT / 16)            // 31250 (E divisible by 16)
#define TILES1  (E1 / 16)               // 25000 (even -> pairs never straddle seam)
#define PAIRS   (TILES / 2)             // 15625 wave work-items (M=32 each)
#define WPB     8                       // waves per block (256 threads)
#define W1P_ELEMS (KCH * NCH * 32 * 16) // 49152 bf16 = 96 KB

// ---------------------------------------------------------------------------
// Prep: convert node features fp32 -> bf16, zero scatter accumulators
// ---------------------------------------------------------------------------
__global__ void gat_prep_convert_zero(const float* __restrict__ x,
                                      __bf16* __restrict__ xbf,
                                      float* __restrict__ accum,
                                      float* __restrict__ rowsum) {
    int i = blockIdx.x * blockDim.x + threadIdx.x;
    if (i >= N_NODES * IN_F) return;
    xbf[i]   = (__bf16)x[i];
    accum[i] = 0.0f;
    if (i < N_NODES) rowsum[i] = 0.0f;
}

// ---------------------------------------------------------------------------
// Prep: pack w_1 [OUT=128][K=384] fp32 into per-lane bf16 B-fragment order.
// B (32x16, 16-bit): lane n (0-15) holds column n, K={0..7,16..23} (2 per VGPR);
// lanes 16-31 hold K={8..15,24..31}.  Fragment = 32 lanes x 16 bf16, per (kc,nc).
// ---------------------------------------------------------------------------
__global__ void gat_prep_pack_w1(const float* __restrict__ w1,
                                 __bf16* __restrict__ w1p) {
    int idx = blockIdx.x * blockDim.x + threadIdx.x;
    if (idx >= W1P_ELEMS) return;
    int j    = idx & 15;
    int lane = (idx >> 4) & 31;
    int nc   = (idx >> 9) & 7;
    int kc   = idx >> 12;
    int n = lane & 15;
    int k = ((lane >> 4) << 3) + (j & 7) + ((j >> 3) << 4);
    w1p[idx] = (__bf16)w1[(nc * 16 + n) * K_TOT + kc * 32 + k];
}

// ---------------------------------------------------------------------------
// Main: per-wave 32-edge pair of tiles (M=32), bf16 WMMA GEMM (32 x 384 x 128)
// with each LDS B fragment reused by two WMMAs, in-register attention logit +
// exp, atomic scatter of weighted messages.
// ---------------------------------------------------------------------------
__global__ __launch_bounds__(256, 1)
void gat_edge_kernel(const __bf16* __restrict__ xbf,
                     const __bf16* __restrict__ w1p,
                     const float*  __restrict__ w2,
                     const long long* __restrict__ el,    // [2][E1]
                     const long long* __restrict__ eln,   // [2][E2]
                     const float*  __restrict__ rel1,     // [E1][128]
                     const float*  __restrict__ rel2,     // [E2][128]
                     float* __restrict__ accum,           // [N][128]
                     float* __restrict__ rowsum) {        // [N]
    __shared__ __align__(16) __bf16 lds_w1[W1P_ELEMS];         // 96 KB
    __shared__ __align__(16) __bf16 lds_a[WPB][2][16 * 32];    // 8 x 2 KB

    // Cooperative load of packed W1 into LDS (block-wide, once).
    {
        const uint4* s = (const uint4*)w1p;
        uint4*       d = (uint4*)lds_w1;
        for (int t = threadIdx.x; t < W1P_ELEMS / 8; t += blockDim.x) d[t] = s[t];
    }
    __syncthreads();

    const int wid  = threadIdx.x >> 5;
    const int lane = threadIdx.x & 31;
    const int pair = blockIdx.x * WPB + wid;
    if (pair >= PAIRS) return;

    const int tile0 = pair * 2;             // both tiles in the same edge segment
    const int row16 = lane & 15;
    const int hi    = lane >> 4;            // lanes 0-15 -> tile0, 16-31 -> tile1

    // Per-lane edge indices: lane<16 holds tile0 row `lane`, lane>=16 tile1 row.
    long long mySrc, myDst;
    const float *relBase0, *relBase1;
    if (tile0 < TILES1) {
        long long eb0 = (long long)tile0 * 16;
        relBase0 = rel1 + eb0 * NRELA;
        relBase1 = rel1 + (eb0 + 16) * NRELA;
        long long e = eb0 + hi * 16 + row16;
        mySrc = el[e];
        myDst = el[E1 + e];
    } else {
        long long eb0 = (long long)(tile0 - TILES1) * 16;
        relBase0 = rel2 + eb0 * NRELA;
        relBase1 = rel2 + (eb0 + 16) * NRELA;
        long long e = eb0 + hi * 16 + row16;
        mySrc = eln[e];
        myDst = eln[E2 + e];
    }

    const int srow  = lane >> 1;   // staging row 0..15 (edge within each tile)
    const int shalf = lane & 1;    // staging half (16 bf16 each)
    const long long gsrc0 = __shfl(mySrc, srow, 32);
    const long long gdst0 = __shfl(myDst, srow, 32);
    const long long gsrc1 = __shfl(mySrc, 16 + srow, 32);
    const long long gdst1 = __shfl(myDst, 16 + srow, 32);

    v8f acc0[NCH], acc1[NCH];
    #pragma unroll
    for (int nc = 0; nc < NCH; ++nc) {
        acc0[nc] = (v8f){0,0,0,0,0,0,0,0};
        acc1[nc] = (v8f){0,0,0,0,0,0,0,0};
    }

    // A-fragment read addressing per ISA 16-bit A layout:
    // lane<16 : K = {0..7} then {16..23};  lane>=16 : K = {8..15} then {24..31}
    __bf16* atile0 = &lds_a[wid][0][0];
    __bf16* atile1 = &lds_a[wid][1][0];
    const __bf16* arow0 = atile0 + row16 * 32 + hi * 8;
    const __bf16* arow1 = atile1 + row16 * 32 + hi * 8;

    #pragma unroll
    for (int kc = 0; kc < KCH; ++kc) {
        __bf16* d0 = atile0 + srow * 32 + shalf * 16;
        __bf16* d1 = atile1 + srow * 32 + shalf * 16;
        if (kc < 8) {
            // gathered bf16 node features: src rows for kc 0..3, dst for 4..7
            const __bf16* g0 = xbf + ((kc < 4) ? gsrc0 : gdst0) * IN_F
                                   + (kc & 3) * 32 + shalf * 16;
            const __bf16* g1 = xbf + ((kc < 4) ? gsrc1 : gdst1) * IN_F
                                   + (kc & 3) * 32 + shalf * 16;
            ((uint4*)d0)[0] = ((const uint4*)g0)[0];
            ((uint4*)d0)[1] = ((const uint4*)g0)[1];
            ((uint4*)d1)[0] = ((const uint4*)g1)[0];
            ((uint4*)d1)[1] = ((const uint4*)g1)[1];
        } else {
            // relation embedding fp32 -> bf16 on the fly
            const float* g0 = relBase0 + (long long)srow * NRELA
                                       + (kc - 8) * 32 + shalf * 16;
            const float* g1 = relBase1 + (long long)srow * NRELA
                                       + (kc - 8) * 32 + shalf * 16;
            #pragma unroll
            for (int j = 0; j < 16; ++j) d0[j] = (__bf16)g0[j];
            #pragma unroll
            for (int j = 0; j < 16; ++j) d1[j] = (__bf16)g1[j];
        }
        __builtin_amdgcn_wave_barrier();
        asm volatile("s_wait_dscnt 0" ::: "memory");

        v8bf a00 = *(const v8bf*)(arow0);
        v8bf a01 = *(const v8bf*)(arow0 + 16);
        v16bf av0 = __builtin_shufflevector(a00, a01, 0, 1, 2, 3, 4, 5, 6, 7,
                                                      8, 9, 10, 11, 12, 13, 14, 15);
        v8bf a10 = *(const v8bf*)(arow1);
        v8bf a11 = *(const v8bf*)(arow1 + 16);
        v16bf av1 = __builtin_shufflevector(a10, a11, 0, 1, 2, 3, 4, 5, 6, 7,
                                                      8, 9, 10, 11, 12, 13, 14, 15);
        const __bf16* bbase = lds_w1 + ((size_t)kc * NCH * 32 + lane) * 16;
        #pragma unroll
        for (int nc = 0; nc < NCH; ++nc) {
            const __bf16* bp = bbase + nc * 32 * 16;
            v8bf b0 = *(const v8bf*)bp;
            v8bf b1 = *(const v8bf*)(bp + 8);
            v16bf bv = __builtin_shufflevector(b0, b1, 0, 1, 2, 3, 4, 5, 6, 7,
                                                       8, 9, 10, 11, 12, 13, 14, 15);
            acc0[nc] = __builtin_amdgcn_wmma_f32_16x16x32_bf16(
                false, av0, false, bv, (short)0, acc0[nc], false, false);
            acc1[nc] = __builtin_amdgcn_wmma_f32_16x16x32_bf16(
                false, av1, false, bv, (short)0, acc1[nc], false, false);
        }
        // DS ops are in-order per wave: next iteration's staging stores cannot
        // bypass this iteration's fragment loads, so no trailing barrier needed.
    }

    // C/D layout: lane n (0-15) holds cols n, VGPR m = row m; lanes 16-31 row m+8.
    const int col = lane & 15;
    float w2v[NCH];
    #pragma unroll
    for (int nc = 0; nc < NCH; ++nc) w2v[nc] = w2[nc * 16 + col];

    float s0[8], s1[8];
    #pragma unroll
    for (int m = 0; m < 8; ++m) {
        float p0 = 0.0f, p1 = 0.0f;
        #pragma unroll
        for (int nc = 0; nc < NCH; ++nc) {
            p0 += acc0[nc][m] * w2v[nc];
            p1 += acc1[nc][m] * w2v[nc];
        }
        s0[m] = p0; s1[m] = p1;
    }
    #pragma unroll
    for (int m = 0; m < 8; ++m) {
        #pragma unroll
        for (int off = 1; off < 16; off <<= 1) {
            s0[m] += __shfl_xor(s0[m], off, 16);
            s1[m] += __shfl_xor(s1[m], off, 16);
        }
    }

    // edge_e = exp(-leaky_relu(s, 0.2))
    float ee0[8], ee1[8];
    #pragma unroll
    for (int m = 0; m < 8; ++m) {
        float v0 = s0[m], v1 = s1[m];
        float l0 = v0 > 0.0f ? v0 : 0.2f * v0;
        float l1 = v1 > 0.0f ? v1 : 0.2f * v1;
        ee0[m] = __expf(-l0);
        ee1[m] = __expf(-l1);
    }

    // src rows this half-wave needs: rows m + 8*hi of each tile
    const int mo = hi * 8;
    long long srcA0[8], srcA1[8];
    #pragma unroll
    for (int m = 0; m < 8; ++m) {
        srcA0[m] = __shfl(mySrc, m + mo, 32);        // tile0 rows on lanes 0-15
        srcA1[m] = __shfl(mySrc, 16 + m + mo, 32);   // tile1 rows on lanes 16-31
    }

    // softmax denominator: lane 0 -> rows 0-7, lane 16 -> rows 8-15 (both tiles)
    if (col == 0) {
        #pragma unroll
        for (int m = 0; m < 8; ++m) {
            atomicAdd(&rowsum[srcA0[m]], ee0[m]);
            atomicAdd(&rowsum[srcA1[m]], ee1[m]);
        }
    }

    // weighted message scatter-sum into L2-resident accumulator
    #pragma unroll
    for (int nc = 0; nc < NCH; ++nc) {
        #pragma unroll
        for (int m = 0; m < 8; ++m) {
            atomicAdd(&accum[srcA0[m] * OUT_F + nc * 16 + col],
                      ee0[m] * acc0[nc][m]);
            atomicAdd(&accum[srcA1[m] * OUT_F + nc * 16 + col],
                      ee1[m] * acc1[nc][m]);
        }
    }
}

// ---------------------------------------------------------------------------
// Finalize: h_prime = accum / rowsum (guard 1e-12), then ELU.
// ---------------------------------------------------------------------------
__global__ void gat_finalize(const float* __restrict__ accum,
                             const float* __restrict__ rowsum,
                             float* __restrict__ out) {
    int i = blockIdx.x * blockDim.x + threadIdx.x;
    if (i >= N_NODES * OUT_F) return;
    int row = i >> 7;
    float r = rowsum[row];
    r = (r == 0.0f) ? 1e-12f : r;
    float h = accum[i] / r;
    out[i] = h > 0.0f ? h : (__expf(h) - 1.0f);
}

// ---------------------------------------------------------------------------
extern "C" void kernel_launch(void* const* d_in, const int* in_sizes, int n_in,
                              void* d_out, int out_size, void* d_ws, size_t ws_size,
                              hipStream_t stream) {
    const float*     input = (const float*)d_in[0];
    const long long* el    = (const long long*)d_in[1];   // int64 [2][E1]
    const float*     rel1  = (const float*)d_in[2];
    const long long* eln   = (const long long*)d_in[3];   // int64 [2][E2]
    const float*     rel2  = (const float*)d_in[4];
    const float*     w1    = (const float*)d_in[5];
    const float*     w2    = (const float*)d_in[6];
    float* out = (float*)d_out;

    char*   ws     = (char*)d_ws;
    size_t  off    = 0;
    __bf16* xbf    = (__bf16*)(ws + off); off += (size_t)N_NODES * IN_F * sizeof(__bf16);
    __bf16* w1p    = (__bf16*)(ws + off); off += (size_t)W1P_ELEMS * sizeof(__bf16);
    float*  accum  = (float*)(ws + off);  off += (size_t)N_NODES * OUT_F * sizeof(float);
    float*  rowsum = (float*)(ws + off);

    gat_prep_convert_zero<<<(N_NODES * IN_F + 255) / 256, 256, 0, stream>>>(
        input, xbf, accum, rowsum);
    gat_prep_pack_w1<<<(W1P_ELEMS + 255) / 256, 256, 0, stream>>>(w1, w1p);
    gat_edge_kernel<<<(PAIRS + WPB - 1) / WPB, 256, 0, stream>>>(
        xbf, w1p, w2, el, eln, rel1, rel2, accum, rowsum);
    gat_finalize<<<(N_NODES * OUT_F + 255) / 256, 256, 0, stream>>>(
        accum, rowsum, out);
}
